// GCN_2800318677196
// MI455X (gfx1250) — matrix operand order repaired and checked by reading, not compiled, not run on hardware
//
#include <hip/hip_runtime.h>
#include <hip/hip_bf16.h>

typedef __attribute__((ext_vector_type(16))) __bf16 v16bf;
typedef __attribute__((ext_vector_type(8)))  float  v8f;

#define NN 50000
#define EE 1600000
#define PP 200000

// ---------------- small utility kernels ----------------

__global__ void fill_val(float* __restrict__ p, float v, int n) {
    int i = blockIdx.x * blockDim.x + threadIdx.x;
    if (i < n) p[i] = v;
}

__global__ void deg_atomic(const int* __restrict__ ei, float* __restrict__ deg, int E) {
    int e = blockIdx.x * blockDim.x + threadIdx.x;
    if (e < E) atomicAdd(&deg[ei[E + e]], 1.0f);   // col = edge_index[1]
}

__global__ void finalize_dis(float* __restrict__ d, int n) {
    int i = blockIdx.x * blockDim.x + threadIdx.x;
    if (i < n) {
        float v = d[i];
        d[i] = (v > 0.0f) ? rsqrtf(v) : 0.0f;
    }
}

// ---------------- WMMA GEMM: H[M,C] = A[M,K] @ W[K,C] (bf16 in, f32 acc) ----------------
// grid.x = M/16 (row tiles), grid.y = C/64 ; 128 threads = 4 waves, wave w -> ctile gy*4+w
__global__ void __launch_bounds__(128)
gemm_bf16_wmma(const float* __restrict__ A, const float* __restrict__ W,
               float* __restrict__ H, int K, int C) {
    const int lane  = threadIdx.x & 31;
    const int wid   = threadIdx.x >> 5;
    const int mtile = blockIdx.x;
    const int ctile = blockIdx.y * 4 + wid;
    const int m     = lane & 15;
    const int nc    = ctile * 16 + m;
    const int kbA   = (lane < 16) ? 0 : 16;    // A frag K-base per lane half
    const int kbB   = (lane < 16) ? 0 : 16;    // B frag K-base per lane half
    const float* arow = A + (size_t)(mtile * 16 + m) * K;

    v8f acc = {};
    for (int k0 = 0; k0 < K; k0 += 32) {
        v16bf a, b;
#pragma unroll
        for (int run = 0; run < 2; ++run) {    // two runs of 8 contiguous K
            const float4* p = (const float4*)(arow + k0 + kbA + run * 8);
            float4 f0 = p[0], f1 = p[1];
            a[run * 8 + 0] = (__bf16)f0.x; a[run * 8 + 1] = (__bf16)f0.y;
            a[run * 8 + 2] = (__bf16)f0.z; a[run * 8 + 3] = (__bf16)f0.w;
            a[run * 8 + 4] = (__bf16)f1.x; a[run * 8 + 5] = (__bf16)f1.y;
            a[run * 8 + 6] = (__bf16)f1.z; a[run * 8 + 7] = (__bf16)f1.w;
        }
        const float* wp = W + (size_t)(k0 + kbB) * C + nc;
#pragma unroll
        for (int j = 0; j < 16; ++j) b[j] = (__bf16)wp[(size_t)j * C];
        acc = __builtin_amdgcn_wmma_f32_16x16x32_bf16(false, a, false, b,
                                                      (short)0, acc, false, false);
    }
    float* hp = H + (size_t)(mtile * 16) * C + nc;
    const int mo = (lane < 16) ? 0 : 8;
#pragma unroll
    for (int r = 0; r < 8; ++r) hp[(size_t)(r + mo) * C] = acc[r];
}

// ---------------- edge scatter: agg[col] += h[row] * dis[row]*dis[col] ----------------
// one thread per (edge, 4-channel group); groups = C/4 = 1<<gshift
__global__ void scatter_edges(const int* __restrict__ ei, const float* __restrict__ dis,
                              const float* __restrict__ h, float* __restrict__ agg,
                              int C, int E, int gshift) {
    long long idx = (long long)blockIdx.x * blockDim.x + threadIdx.x;
    int groups = 1 << gshift;
    long long total = (long long)E * groups;
    if (idx >= total) return;
    int e = (int)(idx >> gshift);
    int g = (int)(idx & (groups - 1));
    int r = ei[e];
    int c = ei[E + e];
    float w = dis[r] * dis[c];
    const float4 hv = *(const float4*)(h + (size_t)r * C + g * 4);
    float* ap = agg + (size_t)c * C + g * 4;
    atomicAdd(ap + 0, hv.x * w);
    atomicAdd(ap + 1, hv.y * w);
    atomicAdd(ap + 2, hv.z * w);
    atomicAdd(ap + 3, hv.w * w);
}

// ---------------- y = relu(agg + h*dis^2 + b), in place; accumulate BN stats ----------------
__global__ void __launch_bounds__(256)
relu_bias_stats(const float* __restrict__ h, const float* __restrict__ dis,
                const float* __restrict__ bias, float* __restrict__ y,
                float* __restrict__ gsum, float* __restrict__ gsq,
                int C, int N, int rowsPerBlock) {
    int ch   = threadIdx.x % C;        // C is 64 or 128
    int sub  = threadIdx.x / C;
    int nsub = 256 / C;
    int r0 = blockIdx.x * rowsPerBlock;
    int r1 = r0 + rowsPerBlock; if (r1 > N) r1 = N;
    float b = bias[ch];
    float s = 0.0f, sq = 0.0f;
    for (int r = r0 + sub; r < r1; r += nsub) {
        size_t off = (size_t)r * C + ch;
        float d = dis[r];
        float v = y[off] + h[off] * d * d + b;   // self-loop folded in
        v = v > 0.0f ? v : 0.0f;
        y[off] = v;
        s += v; sq += v * v;
    }
    __shared__ float ssum[256];
    __shared__ float ssq[256];
    ssum[threadIdx.x] = s;
    ssq[threadIdx.x]  = sq;
    __syncthreads();
    if (sub == 0) {
        for (int k = 1; k < nsub; ++k) { s += ssum[k * C + ch]; sq += ssq[k * C + ch]; }
        atomicAdd(&gsum[ch], s);
        atomicAdd(&gsq[ch], sq);
    }
}

// ---------------- BatchNorm apply (batch stats, biased var) ----------------
__global__ void bn_apply(float* __restrict__ y, const float* __restrict__ g,
                         const float* __restrict__ bt, const float* __restrict__ gsum,
                         const float* __restrict__ gsq, int C, int N) {
    long long idx = (long long)blockIdx.x * blockDim.x + threadIdx.x;
    long long total = (long long)N * C;
    if (idx >= total) return;
    int ch = (int)(idx % C);
    float invN = 1.0f / (float)N;
    float mu  = gsum[ch] * invN;
    float var = gsq[ch] * invN - mu * mu;
    float inv = rsqrtf(var + 1e-5f);
    y[idx] = g[ch] * (y[idx] - mu) * inv + bt[ch];
}

// ---------------- fused edge-score MLP ----------------
// emb[p] = [z[src[p]], z[dst[p]]] (256) ; hidden = relu(emb@fcW1 + fcb1) (128)
// score  = hidden @ fcW2 + fcb2.  One wave = 16 rows; all 8 col-tiles kept in regs.
__global__ void __launch_bounds__(128)
edge_score_wmma(const float* __restrict__ z, const int* __restrict__ src,
                const int* __restrict__ dst, const float* __restrict__ fcW1,
                const float* __restrict__ fcb1, const float* __restrict__ fcW2,
                const float* __restrict__ fcb2, float* __restrict__ out, int P) {
    const int lane  = threadIdx.x & 31;
    const int wid   = threadIdx.x >> 5;
    const int ptile = blockIdx.x * 4 + wid;
    const int p0 = ptile * 16;
    if (p0 >= P) return;                       // wave-uniform guard
    const int m = lane & 15;
    const int p = p0 + m;
    const float* zs = z + (size_t)src[p] * 128;
    const float* zd = z + (size_t)dst[p] * 128;
    const int kbA = (lane < 16) ? 0 : 16;
    const int kbB = (lane < 16) ? 0 : 16;

    v8f acc[8];
#pragma unroll
    for (int i = 0; i < 8; ++i) acc[i] = (v8f){};

    for (int k0 = 0; k0 < 256; k0 += 32) {
        v16bf a;
#pragma unroll
        for (int run = 0; run < 2; ++run) {
            int start = k0 + kbA + run * 8;    // run of 8 never crosses the 128 boundary
            const float4* pp = (const float4*)((start < 128) ? (zs + start)
                                                             : (zd + start - 128));
            float4 f0 = pp[0], f1 = pp[1];
            a[run * 8 + 0] = (__bf16)f0.x; a[run * 8 + 1] = (__bf16)f0.y;
            a[run * 8 + 2] = (__bf16)f0.z; a[run * 8 + 3] = (__bf16)f0.w;
            a[run * 8 + 4] = (__bf16)f1.x; a[run * 8 + 5] = (__bf16)f1.y;
            a[run * 8 + 6] = (__bf16)f1.z; a[run * 8 + 7] = (__bf16)f1.w;
        }
#pragma unroll
        for (int ct = 0; ct < 8; ++ct) {
            v16bf b;
            const float* wp = fcW1 + (size_t)(k0 + kbB) * 128 + ct * 16 + m;
#pragma unroll
            for (int j = 0; j < 16; ++j) b[j] = (__bf16)wp[(size_t)j * 128];
            acc[ct] = __builtin_amdgcn_wmma_f32_16x16x32_bf16(false, a, false, b,
                                                              (short)0, acc[ct], false, false);
        }
    }

    float sc[8];
#pragma unroll
    for (int r = 0; r < 8; ++r) sc[r] = 0.0f;
#pragma unroll
    for (int ct = 0; ct < 8; ++ct) {
        int col = ct * 16 + m;
        float bia = fcb1[col];
        float w2  = fcW2[col];
#pragma unroll
        for (int r = 0; r < 8; ++r) {
            float hv = acc[ct][r] + bia;
            hv = hv > 0.0f ? hv : 0.0f;
            sc[r] += hv * w2;
        }
    }
    // reduce across the 16 lanes of each half (rows M=0..7 low half, M=8..15 high half)
#pragma unroll
    for (int off = 1; off < 16; off <<= 1) {
#pragma unroll
        for (int r = 0; r < 8; ++r) sc[r] += __shfl_xor(sc[r], off, 32);
    }
    if (m == 0) {                               // lanes 0 and 16
        float b2 = fcb2[0];
        int mo = (lane < 16) ? 0 : 8;
#pragma unroll
        for (int r = 0; r < 8; ++r) out[p0 + mo + r] = sc[r] + b2;
    }
}

// ---------------- launch ----------------

extern "C" void kernel_launch(void* const* d_in, const int* in_sizes, int n_in,
                              void* d_out, int out_size, void* d_ws, size_t ws_size,
                              hipStream_t stream) {
    const float* x    = (const float*)d_in[0];
    const int*   ei   = (const int*)  d_in[1];
    const int*   src  = (const int*)  d_in[2];
    const int*   dst  = (const int*)  d_in[3];
    const float* W1   = (const float*)d_in[4];
    const float* b1   = (const float*)d_in[5];
    const float* W2   = (const float*)d_in[6];
    const float* b2   = (const float*)d_in[7];
    const float* W3   = (const float*)d_in[8];
    const float* b3   = (const float*)d_in[9];
    const float* g1   = (const float*)d_in[10];
    const float* bt1  = (const float*)d_in[11];
    const float* g2   = (const float*)d_in[12];
    const float* bt2  = (const float*)d_in[13];
    const float* g3   = (const float*)d_in[14];
    const float* bt3  = (const float*)d_in[15];
    const float* fcW1 = (const float*)d_in[16];
    const float* fcb1 = (const float*)d_in[17];
    const float* fcW2 = (const float*)d_in[18];
    const float* fcb2 = (const float*)d_in[19];
    float* out = (float*)d_out;

    const int N = NN, E = EE, P = PP;

    float* bufA = (float*)d_ws;                    // h scratch   N*128
    float* bufB = bufA + (size_t)N * 128;          // agg/z       N*128
    float* dis  = bufB + (size_t)N * 128;          // deg -> rsqrt(deg), N
    float* gsum = dis + N;                         // 128
    float* gsq  = gsum + 128;                      // 128

    // symmetric normalization: deg (init 1.0 = self loop) -> dis = rsqrt(deg)
    fill_val<<<(N + 255) / 256, 256, 0, stream>>>(dis, 1.0f, N);
    deg_atomic<<<(E + 255) / 256, 256, 0, stream>>>(ei, dis, E);
    finalize_dis<<<(N + 255) / 256, 256, 0, stream>>>(dis, N);

    struct Layer { const float* in; int K; int C; const float* W; const float* b;
                   const float* g; const float* bt; };
    Layer L[3] = {
        { x,    128,  64, W1, b1, g1, bt1 },
        { bufB,  64, 128, W2, b2, g2, bt2 },
        { bufB, 128, 128, W3, b3, g3, bt3 },
    };

    for (int l = 0; l < 3; ++l) {
        const int K = L[l].K, C = L[l].C;
        dim3 gg(N / 16, C / 64);
        gemm_bf16_wmma<<<gg, 128, 0, stream>>>(L[l].in, L[l].W, bufA, K, C);

        hipMemsetAsync(bufB, 0, (size_t)N * C * sizeof(float), stream);
        hipMemsetAsync(gsum, 0, 2 * 128 * sizeof(float), stream);

        const int gshift = (C == 128) ? 5 : 4;          // C/4 groups
        long long tot = (long long)E << gshift;
        scatter_edges<<<(unsigned)((tot + 255) / 256), 256, 0, stream>>>(
            ei, dis, bufA, bufB, C, E, gshift);

        const int rowsPerBlock = 128;
        relu_bias_stats<<<(N + rowsPerBlock - 1) / rowsPerBlock, 256, 0, stream>>>(
            bufA, dis, L[l].b, bufB, gsum, gsq, C, N, rowsPerBlock);

        long long tot2 = (long long)N * C;
        bn_apply<<<(unsigned)((tot2 + 255) / 256), 256, 0, stream>>>(
            bufB, L[l].g, L[l].bt, gsum, gsq, C, N);
    }

    edge_score_wmma<<<P / 64, 128, 0, stream>>>(bufB, src, dst, fcW1, fcb1,
                                                fcW2, fcb2, out, P);
}
